// EncoderBlock_73203422593485
// MI455X (gfx1250) — matrix operand order repaired and checked by reading, not compiled, not run on hardware
//
#include <hip/hip_runtime.h>
#include <hip/hip_bf16.h>

// Problem dims (fixed by the reference)
#define B_   4
#define S_   1024
#define D_   1024
#define H_   16
#define HD_  64
#define DFF_ 4096

typedef __attribute__((ext_vector_type(16))) __bf16 bf16x16;
typedef __attribute__((ext_vector_type(8)))  __bf16 bf16x8;
typedef __attribute__((ext_vector_type(8)))  float  f32x8;
typedef unsigned int v4u __attribute__((ext_vector_type(4)));
typedef int          v8i __attribute__((ext_vector_type(8)));
typedef int          v4i __attribute__((ext_vector_type(4)));

#if __has_builtin(__builtin_amdgcn_tensor_load_to_lds)
#define HAVE_TDM 1
#else
#define HAVE_TDM 0
#endif

static __device__ __forceinline__ f32x8 vzero8() {
  f32x8 v;
#pragma unroll
  for (int i = 0; i < 8; ++i) v[i] = 0.0f;
  return v;
}

static __device__ __forceinline__ bf16x16 cat8(bf16x8 lo, bf16x8 hi) {
  return __builtin_shufflevector(lo, hi, 0,1,2,3,4,5,6,7,8,9,10,11,12,13,14,15);
}

// 16-byte global -> LDS async copy (GVS mode: saddr base + 32-bit vgpr offset).
static __device__ __forceinline__ void async_cp16(uint32_t lds_addr, uint32_t voff,
                                                  uint64_t sbase) {
  asm volatile("global_load_async_to_lds_b128 %0, %1, %2"
               :: "v"(lds_addr), "v"(voff), "s"(sbase) : "memory");
}
static __device__ __forceinline__ void wait_async0() {
  asm volatile("s_wait_asynccnt 0" ::: "memory");
}
static __device__ __forceinline__ uint32_t lds_off(const void* p) {
  return (uint32_t)(uintptr_t)p;   // LDS aperture: low 32 bits = LDS byte address
}

#if HAVE_TDM
// TDM 2D tile load: global (row-major, elements=2B) -> LDS, with LDS row
// padding of 4 DWORDs after every 16 DWORDs (i.e. 64B rows -> 80B LDS rows,
// matching the WMMA fragment layout below).  D# per CDNA5 ISA §8.3/8.4.
static __device__ __forceinline__ void tdm_load_2d(uint32_t lds_addr, uint64_t gaddr,
                                                   uint32_t tdim0, uint32_t tdim1,
                                                   uint32_t tile0, uint32_t tile1,
                                                   uint64_t stride0) {
  v4u g0;
  g0[0] = 1u;                                             // count=1, user D#
  g0[1] = lds_addr;                                       // lds_addr
  g0[2] = (uint32_t)gaddr;                                // global_addr[31:0]
  g0[3] = (uint32_t)((gaddr >> 32) & 0x01FFFFFFu) | 0x80000000u; // addr[56:32] | type=2
  v8i g1;
  g1[0] = (int)((1u << 16) | (1u << 20) | (3u << 22) | (3u << 25));
  //            data_size=2B  pad_enable  interval=16dw  amount=4dw
  g1[1] = (int)((tdim0 & 0xFFFFu) << 16);                 // tensor_dim0[15:0]
  g1[2] = (int)((tdim0 >> 16) | ((tdim1 & 0xFFFFu) << 16));
  g1[3] = (int)((tdim1 >> 16) | (tile0 << 16));           // tile_dim0
  g1[4] = (int)tile1;                                     // tile_dim1 (tile_dim2=0)
  g1[5] = (int)(uint32_t)stride0;                         // tensor_dim0_stride lo
  g1[6] = (int)(uint32_t)((stride0 >> 32) & 0xFFFFu);     // stride hi (dim1_stride=0)
  g1[7] = 0;
  v4i z4 = {0, 0, 0, 0};
#if defined(__clang_major__) && (__clang_major__ >= 23)
  v8i z8 = {0, 0, 0, 0, 0, 0, 0, 0};
  __builtin_amdgcn_tensor_load_to_lds(g0, g1, z4, z4, z8, 0);
#else
  __builtin_amdgcn_tensor_load_to_lds(g0, g1, z4, z4, 0);
#endif
}
#endif

// ---------------------------------------------------------------- converts
__global__ __launch_bounds__(256) void cvt_f32_bf16(const float* __restrict__ in,
                                                    __bf16* __restrict__ out, int n) {
  int i = blockIdx.x * 256 + threadIdx.x;
  if (i < n) out[i] = (__bf16)in[i];
}

// Batched tiled transpose+convert: in[z][K][N] (f32) -> out[z][N][K] (bf16).
__global__ __launch_bounds__(256)
void transpose_f32_bf16(const float* __restrict__ in, __bf16* __restrict__ out,
                        int K, int N) {
  __shared__ __bf16 t[64][68];
  const size_t base = (size_t)blockIdx.z * K * N;
  const int k0 = blockIdx.y * 64, n0 = blockIdx.x * 64;
  const int tid = threadIdx.x;
#pragma unroll
  for (int i = 0; i < 16; ++i) {
    int e = tid + i * 256, r = e >> 6, c = e & 63;
    t[r][c] = (__bf16)in[base + (size_t)(k0 + r) * N + n0 + c];
  }
  __syncthreads();
#pragma unroll
  for (int i = 0; i < 16; ++i) {
    int e = tid + i * 256, r = e >> 6, c = e & 63;
    out[base + (size_t)(n0 + r) * K + k0 + c] = t[c][r];
  }
}

// ---------------------------------------------------------------- GEMM (WMMA bf16)
// C[M,N] = A[M,K] * Bt[N,K]^T + bias, optional exact GELU, f32 and/or bf16 out.
// Both operands are staged as contiguous 2D tiles.  Preferred path: the Tensor
// Data Mover (one tensor_load_to_lds per tile, issued by wave 0, fenced with
// TENSORcnt + workgroup barrier), double-buffered so DMA overlaps WMMA.
// Fallback (no TDM builtin): per-lane GLOBAL_LOAD_ASYNC_TO_LDS_B128.
template<int BM, int BN, bool OUTF32, bool OUTBF16, bool GELU_, bool HEAD>
__global__ __launch_bounds__(256)
void gemm_bf16_wmma(const __bf16* __restrict__ A, const __bf16* __restrict__ Bt,
                    const float* __restrict__ bias,
                    float* __restrict__ Cf, __bf16* __restrict__ Cb,
                    int M, int N, int K) {
  constexpr int BK  = 32;
  constexpr int LDT = BK + 8;          // 80-byte row stride (TDM pad reproduces this)
  constexpr int MI  = (BM / 2) / 16;   // waves 2x4
  constexpr int NI  = (BN / 4) / 16;
  constexpr int ACH = BM / 64;         // async fallback: 16B chunks per thread
  constexpr int BCH = BN / 64;

  __shared__ __align__(16) __bf16 sA[2][BM][LDT];
  __shared__ __align__(16) __bf16 sB[2][BN][LDT];

  const int tid  = threadIdx.x;
  const int lane = tid & 31, wid = tid >> 5;
  const int wm   = wid >> 2, wn = wid & 3;
  const int m0   = blockIdx.y * BM;
  const int n0   = blockIdx.x * BN;

  const __bf16* Bb    = Bt;
  const float*  biasb = bias;
  if (HEAD) { Bb += (size_t)blockIdx.z * (size_t)N * K; biasb += blockIdx.z * N; }
  const uint64_t aBase = (uint64_t)A;
  const uint64_t bBase = (uint64_t)Bb;

  const uint32_t sA0 = lds_off(&sA[0][0][0]);
  const uint32_t sB0 = lds_off(&sB[0][0][0]);
  constexpr uint32_t ABUF = BM * LDT * 2;
  constexpr uint32_t BBUF = BN * LDT * 2;

  f32x8 acc[MI][NI];
#pragma unroll
  for (int mi = 0; mi < MI; ++mi)
#pragma unroll
    for (int ni = 0; ni < NI; ++ni) acc[mi][ni] = vzero8();

  const int arow = lane & 15;
  const int klo  = (lane < 16) ? 0 : 8;   // A frag: K split per lane half
  const int kb   = (lane < 16) ? 0 : 16;  // unused; B frag also uses klo (same k order as A)

  auto stage = [&](int buf, int k0) {
#if HAVE_TDM
    if (wid == 0) {
      tdm_load_2d(sA0 + buf * ABUF, aBase + ((uint64_t)(size_t)m0 * K + k0) * 2,
                  (uint32_t)K, (uint32_t)M, BK, BM, (uint64_t)K);
      tdm_load_2d(sB0 + buf * BBUF, bBase + ((uint64_t)(size_t)n0 * K + k0) * 2,
                  (uint32_t)K, (uint32_t)N, BK, BN, (uint64_t)K);
    }
#else
#pragma unroll
    for (int it = 0; it < ACH; ++it) {
      int cidx = tid + it * 256, r = cidx >> 2, cc = cidx & 3;
      uint32_t voff = (uint32_t)((((m0 + r) * K) + k0 + cc * 8) * 2);
      async_cp16(sA0 + buf * ABUF + r * (LDT * 2) + cc * 16, voff, aBase);
    }
#pragma unroll
    for (int it = 0; it < BCH; ++it) {
      int cidx = tid + it * 256, r = cidx >> 2, cc = cidx & 3;
      uint32_t voff = (uint32_t)((((n0 + r) * K) + k0 + cc * 8) * 2);
      async_cp16(sB0 + buf * BBUF + r * (LDT * 2) + cc * 16, voff, bBase);
    }
#endif
  };
  auto stage_wait = [&]() {
#if HAVE_TDM
    if (wid == 0) __builtin_amdgcn_s_wait_tensorcnt(0);
#else
    wait_async0();
#endif
  };

  stage(0, 0);
  const int nkt = K / BK;
  for (int t = 0; t < nkt; ++t) {
    stage_wait();                  // current tile resident (issuing wave)
    __syncthreads();               // ... published to the whole workgroup
    const int buf = t & 1;
    if (t + 1 < nkt) stage(buf ^ 1, (t + 1) * BK);   // DMA overlaps compute

    bf16x16 af[MI];
#pragma unroll
    for (int mi = 0; mi < MI; ++mi) {
      const __bf16* rp = &sA[buf][wm * (BM / 2) + mi * 16 + arow][0];
      af[mi] = cat8(*(const bf16x8*)(rp + klo), *(const bf16x8*)(rp + klo + 16));
    }
#pragma unroll
    for (int ni = 0; ni < NI; ++ni) {
      const __bf16* rp = &sB[buf][wn * (BN / 4) + ni * 16 + (lane & 15)][0];
      bf16x16 bf = cat8(*(const bf16x8*)(rp + klo), *(const bf16x8*)(rp + klo + 16));
#pragma unroll
      for (int mi = 0; mi < MI; ++mi)
        acc[mi][ni] = __builtin_amdgcn_wmma_f32_16x16x32_bf16(
            false, af[mi], false, bf, (short)0, acc[mi][ni], false, false);
    }
    __syncthreads();               // done reading buf before it is refilled
  }

  const int rofs = (lane & 16) ? 8 : 0;
#pragma unroll
  for (int mi = 0; mi < MI; ++mi)
#pragma unroll
    for (int ni = 0; ni < NI; ++ni)
#pragma unroll
      for (int r = 0; r < 8; ++r) {
        int gm = m0 + wm * (BM / 2) + mi * 16 + r + rofs;
        int gn = n0 + wn * (BN / 4) + ni * 16 + (lane & 15);
        float v = acc[mi][ni][r] + biasb[gn];
        if (GELU_) v = 0.5f * v * (1.0f + erff(v * 0.70710678f));
        size_t idx;
        if (HEAD) {
          int b = gm / S_, s = gm % S_;
          idx = (((size_t)b * H_ + blockIdx.z) * S_ + s) * HD_ + gn;
        } else {
          idx = (size_t)gm * N + gn;
        }
        if (OUTF32)  Cf[idx] = v;
        if (OUTBF16) Cb[idx] = (__bf16)v;
      }
}

// ---------------------------------------------------------------- flash attention
// Q,K,V: [B,H,S,HD] bf16. Per block: one (b,h), 128 query rows (16 per wave).
// Q and K tiles staged by GLOBAL_LOAD_ASYNC_TO_LDS; V needs an element
// transpose (ds path). Online softmax; O written to [B,S,D].
__global__ __launch_bounds__(256)
void flash_attn_wmma(const __bf16* __restrict__ Q, const __bf16* __restrict__ Kk,
                     const __bf16* __restrict__ Vv, __bf16* __restrict__ O) {
  constexpr int LD = 72;               // 144-byte row stride (16B aligned)
  __shared__ __align__(16) __bf16 sQ[128][LD];
  __shared__ __align__(16) __bf16 sK[64][LD];    // sK[key][dim]
  __shared__ __align__(16) __bf16 sVt[64][LD];   // sVt[dim][key]
  __shared__ __align__(16) __bf16 sP[8][16][LD]; // per-wave probability tile

  const int tid = threadIdx.x, lane = tid & 31, wid = tid >> 5;
  const int m0 = blockIdx.x * 128;
  const size_t base = (size_t)blockIdx.y * S_ * HD_;
  const uint64_t qBase = (uint64_t)(Q + base);
  const uint64_t kBase = (uint64_t)(Kk + base);
  const uint32_t sQ0 = lds_off(&sQ[0][0]);
  const uint32_t sK0 = lds_off(&sK[0][0]);

#pragma unroll
  for (int it = 0; it < 2; ++it) {     // async-load whole Q block
    int cidx = tid + it * 256, r = cidx >> 2, cc = cidx & 3;
    uint32_t voff = (uint32_t)((((m0 + r) * HD_) + cc * 8) * 2);
    async_cp16(sQ0 + r * (LD * 2) + cc * 16, voff, qBase);
  }

  f32x8 oacc[4];
#pragma unroll
  for (int nd = 0; nd < 4; ++nd) oacc[nd] = vzero8();
  float mrow[8], lrow[8];
#pragma unroll
  for (int r = 0; r < 8; ++r) { mrow[r] = -3.0e38f; lrow[r] = 0.0f; }

  const int arow = lane & 15;
  const int klo  = (lane < 16) ? 0 : 8;
  const int kb   = (lane < 16) ? 0 : 16;
  const int prow = (lane & 16) ? 8 : 0;
  const float scale = 0.03125f;        // 1/sqrt(D)=1/32 (reference uses d_model)
  const float LOG2E = 1.44269504f;

  for (int kt = 0; kt < S_ / 64; ++kt) {
    __syncthreads();                   // previous tile fully consumed
    {                                  // async K tile (256 x 16B chunks)
      int r = tid >> 2, cc = tid & 3;
      uint32_t voff = (uint32_t)((((kt * 64 + r) * HD_) + cc * 8) * 2);
      async_cp16(sK0 + r * (LD * 2) + cc * 16, voff, kBase);
    }
#pragma unroll
    for (int it = 0; it < 16; ++it) {  // V transpose via ds path
      int e = tid + it * 256, r = e >> 6, c = e & 63;
      sVt[c][r] = Vv[base + (size_t)(kt * 64 + r) * HD_ + c];
    }
    wait_async0();
    __syncthreads();

    // S_tile = Q(16xHD) * K^T(HDx64)
    f32x8 sacc[4];
#pragma unroll
    for (int nk = 0; nk < 4; ++nk) sacc[nk] = vzero8();
#pragma unroll
    for (int ks = 0; ks < 64; ks += 32) {
      const __bf16* qp = &sQ[wid * 16 + arow][ks];
      bf16x16 aq = cat8(*(const bf16x8*)(qp + klo), *(const bf16x8*)(qp + klo + 16));
#pragma unroll
      for (int nk = 0; nk < 4; ++nk) {
        const __bf16* kp = &sK[nk * 16 + arow][ks];
        bf16x16 bk = cat8(*(const bf16x8*)(kp + klo), *(const bf16x8*)(kp + klo + 16));
        sacc[nk] = __builtin_amdgcn_wmma_f32_16x16x32_bf16(
            false, aq, false, bk, (short)0, sacc[nk], false, false);
      }
    }

    // online softmax across this tile's 64 keys (row-wise, 16-lane groups)
#pragma unroll
    for (int r = 0; r < 8; ++r) {
      float s0 = sacc[0][r] * scale, s1 = sacc[1][r] * scale;
      float s2 = sacc[2][r] * scale, s3 = sacc[3][r] * scale;
      float mx = fmaxf(fmaxf(s0, s1), fmaxf(s2, s3));
#pragma unroll
      for (int o = 8; o >= 1; o >>= 1) mx = fmaxf(mx, __shfl_xor(mx, o, 32));
      float mnew  = fmaxf(mrow[r], mx);
      float alpha = exp2f((mrow[r] - mnew) * LOG2E);
      float p0 = exp2f((s0 - mnew) * LOG2E), p1 = exp2f((s1 - mnew) * LOG2E);
      float p2 = exp2f((s2 - mnew) * LOG2E), p3 = exp2f((s3 - mnew) * LOG2E);
      float rs = p0 + p1 + p2 + p3;
#pragma unroll
      for (int o = 8; o >= 1; o >>= 1) rs += __shfl_xor(rs, o, 32);
      lrow[r] = lrow[r] * alpha + rs;
      mrow[r] = mnew;
#pragma unroll
      for (int nd = 0; nd < 4; ++nd) oacc[nd][r] *= alpha;
      __bf16* pp = &sP[wid][r + prow][lane & 15];
      pp[0] = (__bf16)p0; pp[16] = (__bf16)p1; pp[32] = (__bf16)p2; pp[48] = (__bf16)p3;
    }
    asm volatile("s_wait_dscnt 0" ::: "memory");  // per-wave LDS round-trip

    // O += P(16x64) * V(64xHD)
#pragma unroll
    for (int ks = 0; ks < 64; ks += 32) {
      const __bf16* pr = &sP[wid][arow][ks];
      bf16x16 ap = cat8(*(const bf16x8*)(pr + klo), *(const bf16x8*)(pr + klo + 16));
#pragma unroll
      for (int nd = 0; nd < 4; ++nd) {
        const __bf16* vp = &sVt[nd * 16 + arow][ks];
        bf16x16 bv = cat8(*(const bf16x8*)(vp + klo), *(const bf16x8*)(vp + klo + 16));
        oacc[nd] = __builtin_amdgcn_wmma_f32_16x16x32_bf16(
            false, ap, false, bv, (short)0, oacc[nd], false, false);
      }
    }
  }

  const int b = blockIdx.y / H_, h = blockIdx.y % H_;
#pragma unroll
  for (int nd = 0; nd < 4; ++nd)
#pragma unroll
    for (int r = 0; r < 8; ++r) {
      int grow = m0 + wid * 16 + r + prow;
      int dcol = h * HD_ + nd * 16 + (lane & 15);
      float v = oacc[nd][r] / lrow[r];
      O[((size_t)b * S_ + grow) * D_ + dcol] = (__bf16)v;
    }
}

// ---------------------------------------------------------------- fused add + LayerNorm
__global__ __launch_bounds__(256)
void add_layernorm(const float* __restrict__ X, const float* __restrict__ Y,
                   const float* __restrict__ gamma, const float* __restrict__ beta,
                   float* __restrict__ outF, __bf16* __restrict__ outB) {
  __shared__ float red[2][8];
  __shared__ float stats[2];
  const int row = blockIdx.x;
  const int tid = threadIdx.x, lane = tid & 31, wid = tid >> 5;
  float v[4], s = 0.0f, ss = 0.0f;
#pragma unroll
  for (int i = 0; i < 4; ++i) {
    int c = tid + i * 256;
    float x = X[(size_t)row * D_ + c] + Y[(size_t)row * D_ + c];
    v[i] = x; s += x; ss += x * x;
  }
#pragma unroll
  for (int o = 16; o >= 1; o >>= 1) { s += __shfl_xor(s, o, 32); ss += __shfl_xor(ss, o, 32); }
  if (lane == 0) { red[0][wid] = s; red[1][wid] = ss; }
  __syncthreads();
  if (tid == 0) {
    float ts = 0, tss = 0;
    for (int i = 0; i < 8; ++i) { ts += red[0][i]; tss += red[1][i]; }
    float mu = ts / (float)D_;
    stats[0] = mu;
    stats[1] = rsqrtf(tss / (float)D_ - mu * mu + 1e-5f);
  }
  __syncthreads();
  float mu = stats[0], rs = stats[1];
#pragma unroll
  for (int i = 0; i < 4; ++i) {
    int c = tid + i * 256;
    float y = (v[i] - mu) * rs * gamma[c] + beta[c];
    outF[(size_t)row * D_ + c] = y;
    if (outB) outB[(size_t)row * D_ + c] = (__bf16)y;
  }
}

// ---------------------------------------------------------------- launch
extern "C" void kernel_launch(void* const* d_in, const int* in_sizes, int n_in,
                              void* d_out, int out_size, void* d_ws, size_t ws_size,
                              hipStream_t stream) {
  (void)in_sizes; (void)n_in; (void)out_size; (void)ws_size;
  const float* x    = (const float*)d_in[0];
  const float* Wk   = (const float*)d_in[1];  const float* bk  = (const float*)d_in[2];
  const float* Wq   = (const float*)d_in[3];  const float* bq  = (const float*)d_in[4];
  const float* Wv   = (const float*)d_in[5];  const float* bv  = (const float*)d_in[6];
  const float* Whq  = (const float*)d_in[7];  const float* bhq = (const float*)d_in[8];
  const float* Whk  = (const float*)d_in[9];  const float* bhk = (const float*)d_in[10];
  const float* Whv  = (const float*)d_in[11]; const float* bhv = (const float*)d_in[12];
  const float* Wo   = (const float*)d_in[13]; const float* bo  = (const float*)d_in[14];
  const float* W1   = (const float*)d_in[15]; const float* b1  = (const float*)d_in[16];
  const float* W2   = (const float*)d_in[17]; const float* b2  = (const float*)d_in[18];
  const float* gamma = (const float*)d_in[19];
  const float* beta  = (const float*)d_in[20];

  const size_t DD  = (size_t)D_ * D_;
  const size_t BSD = (size_t)B_ * S_ * D_;
  const size_t DF1 = (size_t)D_ * DFF_;
  const size_t BSF = (size_t)B_ * S_ * DFF_;

  char* p = (char*)d_ws;
  auto abf = [&](size_t n) { __bf16* r = (__bf16*)p; p += (n * 2 + 255) & ~(size_t)255; return r; };
  auto af32 = [&](size_t n) { float* r = (float*)p; p += (n * 4 + 255) & ~(size_t)255; return r; };

  // all weight buffers hold the TRANSPOSED bf16 weights ([N][K])
  __bf16 *wX   = abf(BSD);
  __bf16 *wWk  = abf(DD), *wWq = abf(DD), *wWv = abf(DD);
  __bf16 *wWhq = abf(DD), *wWhk = abf(DD), *wWhv = abf(DD);
  __bf16 *wWo  = abf(DD), *wW1 = abf(DF1), *wW2 = abf(DF1);
  __bf16 *wKo  = abf(BSD), *wQo = abf(BSD), *wVo = abf(BSD);
  __bf16 *wQh  = abf(BSD), *wKh = abf(BSD), *wVh = abf(BSD);
  __bf16 *wO   = abf(BSD), *wR1b = abf(BSD), *wFF1 = abf(BSF);
  float  *wAttn = af32(BSD), *wR1 = af32(BSD), *wFF2 = af32(BSD);

  const int M = B_ * S_;  // 4096
  dim3 blk(256);

  cvt_f32_bf16<<<(int)((BSD + 255) / 256), blk, 0, stream>>>(x, wX, (int)BSD);
  transpose_f32_bf16<<<dim3(16, 16, 1),  blk, 0, stream>>>(Wk,  wWk,  D_, D_);
  transpose_f32_bf16<<<dim3(16, 16, 1),  blk, 0, stream>>>(Wq,  wWq,  D_, D_);
  transpose_f32_bf16<<<dim3(16, 16, 1),  blk, 0, stream>>>(Wv,  wWv,  D_, D_);
  transpose_f32_bf16<<<dim3(1, 16, H_),  blk, 0, stream>>>(Whq, wWhq, D_, HD_);
  transpose_f32_bf16<<<dim3(1, 16, H_),  blk, 0, stream>>>(Whk, wWhk, D_, HD_);
  transpose_f32_bf16<<<dim3(1, 16, H_),  blk, 0, stream>>>(Whv, wWhv, D_, HD_);
  transpose_f32_bf16<<<dim3(16, 16, 1),  blk, 0, stream>>>(Wo,  wWo,  D_, D_);
  transpose_f32_bf16<<<dim3(64, 16, 1),  blk, 0, stream>>>(W1,  wW1,  D_, DFF_);
  transpose_f32_bf16<<<dim3(16, 64, 1),  blk, 0, stream>>>(W2,  wW2,  DFF_, D_);

  // outer projections (bf16 out, feed head projections)
  gemm_bf16_wmma<128,128,false,true,false,false>
      <<<dim3(D_/128, M/128), blk, 0, stream>>>(wX, wWk, bk, nullptr, wKo, M, D_, D_);
  gemm_bf16_wmma<128,128,false,true,false,false>
      <<<dim3(D_/128, M/128), blk, 0, stream>>>(wX, wWq, bq, nullptr, wQo, M, D_, D_);
  gemm_bf16_wmma<128,128,false,true,false,false>
      <<<dim3(D_/128, M/128), blk, 0, stream>>>(wX, wWv, bv, nullptr, wVo, M, D_, D_);

  // per-head projections — NOTE reference swap: MHA Q stream <- k_proj, K <- q_proj
  gemm_bf16_wmma<128,64,false,true,false,true>
      <<<dim3(1, M/128, H_), blk, 0, stream>>>(wKo, wWhq, bhq, nullptr, wQh, M, HD_, D_);
  gemm_bf16_wmma<128,64,false,true,false,true>
      <<<dim3(1, M/128, H_), blk, 0, stream>>>(wQo, wWhk, bhk, nullptr, wKh, M, HD_, D_);
  gemm_bf16_wmma<128,64,false,true,false,true>
      <<<dim3(1, M/128, H_), blk, 0, stream>>>(wVo, wWhv, bhv, nullptr, wVh, M, HD_, D_);

  // fused attention (no materialized scores)
  flash_attn_wmma<<<dim3(S_/128, B_*H_), blk, 0, stream>>>(wQh, wKh, wVh, wO);

  // output projection (f32 out for residual)
  gemm_bf16_wmma<128,128,true,false,false,false>
      <<<dim3(D_/128, M/128), blk, 0, stream>>>(wO, wWo, bo, wAttn, nullptr, M, D_, D_);

  // r1 = LN(x + attn)
  add_layernorm<<<M, blk, 0, stream>>>(x, wAttn, gamma, beta, wR1, wR1b);

  // FFN
  gemm_bf16_wmma<128,128,false,true,true,false>
      <<<dim3(DFF_/128, M/128), blk, 0, stream>>>(wR1b, wW1, b1, nullptr, wFF1, M, DFF_, D_);
  gemm_bf16_wmma<128,128,true,false,false,false>
      <<<dim3(D_/128, M/128), blk, 0, stream>>>(wFF1, wW2, b2, wFF2, nullptr, M, D_, DFF_);

  // out = LN(r1 + ff2)
  add_layernorm<<<M, blk, 0, stream>>>(wR1, wFF2, gamma, beta, (float*)d_out, nullptr);
}
